// MoELTSMemory_8581344657504
// MI455X (gfx1250) — compile-verified, matrix operand order checked by reading, not run on hardware
//
#include <hip/hip_runtime.h>

typedef unsigned short u16;
typedef __attribute__((ext_vector_type(16))) __bf16 v16bf;
typedef __attribute__((ext_vector_type(8)))  float  v8f;
typedef __attribute__((ext_vector_type(8)))  unsigned short ushort8;
typedef __attribute__((ext_vector_type(4)))  int    v4i;

#define TOK   8192      // B*S
#define DMOD  2048
#define DMEM  512
#define NEXP  8
#define CAPS  1024
#define CH    128       // expert kernel memory-slot chunk

#define AS1 __attribute__((address_space(1)))
#define AS3 __attribute__((address_space(3)))

#if defined(__gfx1250__) && __has_builtin(__builtin_amdgcn_global_load_async_to_lds_b128) && __has_builtin(__builtin_amdgcn_s_wait_asynccnt)
#define HAVE_ASYNC_LDS 1
#else
#define HAVE_ASYNC_LDS 0
#endif

// ---------- bf16 helpers ----------
__device__ __forceinline__ u16 f2bf(float x) {
  unsigned int u = __float_as_uint(x);
  u += 0x7fffu + ((u >> 16) & 1u);          // round-to-nearest-even
  return (u16)(u >> 16);
}
__device__ __forceinline__ float bf2f(u16 h) {
  return __uint_as_float(((unsigned int)h) << 16);
}

// 16B global -> LDS copy: async-to-LDS when available, else through VGPRs.
__device__ __forceinline__ void cp16(u16* l, const u16* g) {
#if HAVE_ASYNC_LDS
  __builtin_amdgcn_global_load_async_to_lds_b128((AS1 v4i*)g, (AS3 v4i*)l, 0, 0);
#else
  *(ushort8*)l = *(const ushort8*)g;
#endif
}
__device__ __forceinline__ void async_fence() {
#if HAVE_ASYNC_LDS
  __builtin_amdgcn_s_wait_asynccnt(0);
#endif
}

union Frag16 { v16bf bf; struct { ushort8 lo, hi; } u; };

// Load a 16-bit WMMA A/B fragment from a row-major [rows][ld] bf16 buffer.
// 16-bit layout: lane half (>=16) shifts K by 8; elements 0-7 = K..K+7,
// elements 8-15 = K+16..K+23 (two contiguous 16B groups).
__device__ __forceinline__ v16bf load_frag(const u16* base, int ld, int row,
                                           int kbase, int lane) {
  const u16* p = base + (size_t)row * ld + kbase + ((lane & 16) ? 8 : 0);
  Frag16 f;
  f.u.lo = *(const ushort8*)(p);
  f.u.hi = *(const ushort8*)(p + 16);
  return f.bf;
}

__device__ __forceinline__ v8f wmma_bf16(v16bf a, v16bf b, v8f c) {
  return __builtin_amdgcn_wmma_f32_16x16x32_bf16(false, a, false, b,
                                                 (short)0, c, false, false);
}

// ---------- elementwise f32 -> bf16 ----------
__global__ void cvt_bf16_kernel(const float* __restrict__ src,
                                u16* __restrict__ dst, int n) {
  for (int i = blockIdx.x * blockDim.x + threadIdx.x; i < n;
       i += gridDim.x * blockDim.x)
    dst[i] = f2bf(src[i]);
}

// ---------- memories: [E][CAP][DMEM] f32 -> [E][DMEM][CAP] bf16 ----------
__global__ void transpose_mem_kernel(const float* __restrict__ mem,
                                     u16* __restrict__ memT) {
  __shared__ float tile[32][33];
  int e = blockIdx.z;
  int d0 = blockIdx.x * 32, s0 = blockIdx.y * 32;
  int x = threadIdx.x & 31, y = threadIdx.x >> 5;   // 256 threads: 32 x 8
  const float* src = mem + ((size_t)e * CAPS + s0) * DMEM + d0;
#pragma unroll
  for (int i = 0; i < 4; i++) {
    int row = y + i * 8;
    tile[row][x] = src[(size_t)row * DMEM + x];
  }
  __syncthreads();
  u16* dst = memT + ((size_t)e * DMEM + d0) * CAPS + s0;
#pragma unroll
  for (int i = 0; i < 4; i++) {
    int row = y + i * 8;
    dst[(size_t)row * CAPS + x] = f2bf(tile[x][row]);
  }
}

// ---------- counters init ----------
__global__ void init_counters_kernel(int* c) {
  if (threadIdx.x < 32) c[threadIdx.x] = 0;   // cnt[8], fill[8], disp[8], offsets[8]
}

// ---------- router: one wave per token ----------
__global__ void router_kernel(const float* __restrict__ hidden,
                              const float* __restrict__ rw,
                              int* __restrict__ top_idx,
                              float* __restrict__ top_w,
                              float* __restrict__ weights8,
                              int* __restrict__ disp,
                              int* __restrict__ cnt) {
  int wave = threadIdx.x >> 5;
  int lane = threadIdx.x & 31;
  int t = blockIdx.x * 8 + wave;
  if (t >= TOK) return;

  const float* h = hidden + (size_t)t * DMOD;
  float acc[NEXP];
#pragma unroll
  for (int e = 0; e < NEXP; e++) acc[e] = 0.f;

  for (int i = lane * 4; i < DMOD; i += 128) {
    float4 hv = *(const float4*)(h + i);
#pragma unroll
    for (int e = 0; e < NEXP; e++) {
      float4 wv = *(const float4*)(rw + e * DMOD + i);
      acc[e] += hv.x * wv.x + hv.y * wv.y + hv.z * wv.z + hv.w * wv.w;
    }
  }
#pragma unroll
  for (int e = 0; e < NEXP; e++)
#pragma unroll
    for (int off = 16; off > 0; off >>= 1)
      acc[e] += __shfl_down(acc[e], off, 32);

  if (lane == 0) {
    float m = acc[0];
#pragma unroll
    for (int e = 1; e < NEXP; e++) m = fmaxf(m, acc[e]);
    float p[NEXP], s = 0.f;
#pragma unroll
    for (int e = 0; e < NEXP; e++) { p[e] = __expf(acc[e] - m); s += p[e]; }
    float inv = 1.f / s;
    int i0 = 0, i1 = -1;
    float w0 = -1.f, w1 = -1.f;
#pragma unroll
    for (int e = 0; e < NEXP; e++) {
      p[e] *= inv;
      weights8[t * NEXP + e] = p[e];
      if (p[e] > w0) { w1 = w0; i1 = i0; w0 = p[e]; i0 = e; }
      else if (p[e] > w1) { w1 = p[e]; i1 = e; }
    }
    atomicAdd(&disp[i0], 1);
    atomicAdd(&cnt[i0], 1);
    atomicAdd(&cnt[i1], 1);
    float rn = 1.f / (w0 + w1 + 1e-8f);
    top_idx[t * 2 + 0] = i0;  top_idx[t * 2 + 1] = i1;
    top_w[t * 2 + 0] = w0 * rn;  top_w[t * 2 + 1] = w1 * rn;
  }
}

// ---------- deterministic loss + expert offsets ----------
__global__ void scan_loss_kernel(const float* __restrict__ weights8,
                                 const int* __restrict__ disp,
                                 const int* __restrict__ cnt,
                                 int* __restrict__ offsets,
                                 float* __restrict__ loss_out) {
  __shared__ float red[256];
  __shared__ float prob[NEXP];
  int tid = threadIdx.x;
  for (int e = 0; e < NEXP; e++) {
    float part = 0.f;
    for (int t = tid; t < TOK; t += 256) part += weights8[t * NEXP + e];
    red[tid] = part;
    __syncthreads();
    for (int s = 128; s > 0; s >>= 1) {
      if (tid < s) red[tid] += red[tid + s];
      __syncthreads();
    }
    if (tid == 0) prob[e] = red[0];
    __syncthreads();
  }
  if (tid == 0) {
    float loss = 0.f;
    for (int e = 0; e < NEXP; e++)
      loss += ((float)disp[e] / (float)TOK) * (prob[e] / (float)TOK);
    loss_out[0] = loss * (float)NEXP;
    int run = 0;
    for (int e = 0; e < NEXP; e++) { offsets[e] = run; run += cnt[e]; }
  }
}

// ---------- scatter tokens into per-expert lists ----------
__global__ void scatter_kernel(const int* __restrict__ top_idx,
                               const float* __restrict__ top_w,
                               const int* __restrict__ offsets,
                               int* __restrict__ fill,
                               int* __restrict__ a_tok,
                               float* __restrict__ a_w,
                               int* __restrict__ a_slot) {
  int t = blockIdx.x * blockDim.x + threadIdx.x;
  if (t >= TOK) return;
#pragma unroll
  for (int k = 0; k < 2; k++) {
    int e = top_idx[t * 2 + k];
    int pos = atomicAdd(&fill[e], 1);
    int idx = offsets[e] + pos;
    a_tok[idx] = t;
    a_w[idx] = top_w[t * 2 + k];
    a_slot[idx] = t * 2 + k;
  }
}

// ---------- generic bf16 WMMA GEMM: C[M,N] = A[M,K] * B[N,K]^T + bias ----------
// block = 128 threads (4 waves); block tile 32 x 256; wave tile 32 x 64.
// Each global B-fragment feeds two WMMAs (two M-tiles); B loads are
// software-pipelined one tile ahead of the consuming WMMA.
template <bool OUT_BF16>
__global__ void gemm_bf16_kernel(const u16* __restrict__ A,
                                 const u16* __restrict__ Bm,
                                 const float* __restrict__ bias,
                                 void* __restrict__ C,
                                 int M, int N, int K) {
  __shared__ __align__(16) u16 As[32 * 72];
  int tid = threadIdx.x;
  int wave = tid >> 5, lane = tid & 31;
  int row0 = blockIdx.y * 32;
  int col0 = blockIdx.x * 256 + wave * 64;

  v8f acc[2][4];
#pragma unroll
  for (int rt = 0; rt < 2; rt++)
#pragma unroll
    for (int nt = 0; nt < 4; nt++) acc[rt][nt] = (v8f){0,0,0,0,0,0,0,0};

  for (int kc = 0; kc < K; kc += 64) {
    __syncthreads();
    // stage A tile 32x64 (two 16B async copies per thread)
#pragma unroll
    for (int i = 0; i < 2; i++) {
      int idx = tid + i * 128;
      int r = idx >> 3, seg = idx & 7;
      cp16(&As[r * 72 + seg * 8],
           A + (size_t)(row0 + r) * K + kc + seg * 8);
    }
    async_fence();
    __syncthreads();
#pragma unroll
    for (int kk = 0; kk < 2; kk++) {
      v16bf a0 = load_frag(As, 72, (lane & 15), kk * 32, lane);
      v16bf a1 = load_frag(As, 72, 16 + (lane & 15), kk * 32, lane);
      v16bf b = load_frag(Bm, K, col0 + (lane & 15), kc + kk * 32, lane);
#pragma unroll
      for (int nt = 0; nt < 4; nt++) {
        v16bf bn = b;
        if (nt < 3)
          bn = load_frag(Bm, K, col0 + (nt + 1) * 16 + (lane & 15),
                         kc + kk * 32, lane);
        acc[0][nt] = wmma_bf16(a0, b, acc[0][nt]);
        acc[1][nt] = wmma_bf16(a1, b, acc[1][nt]);
        b = bn;
      }
    }
  }
#pragma unroll
  for (int rt = 0; rt < 2; rt++) {
    int rbase = row0 + rt * 16 + ((lane & 16) ? 8 : 0);
#pragma unroll
    for (int nt = 0; nt < 4; nt++) {
      int c = col0 + nt * 16 + (lane & 15);
      float badd = bias ? bias[c] : 0.f;
#pragma unroll
      for (int v = 0; v < 8; v++) {
        int r = rbase + v;
        float val = acc[rt][nt][v] + badd;
        if (OUT_BF16) ((u16*)C)[(size_t)r * N + c] = f2bf(val);
        else          ((float*)C)[(size_t)r * N + c] = val;
      }
    }
  }
}

// ---------- expert flash-attention over memory bank ----------
// grid = (maxTiles, NEXP), block = 128 (4 waves), tile = 16 assignments.
// Chunk = 128 slots; each wave computes two QK^T slot-tiles (A-frag reuse x2).
// QK^T B-frags stream from row-major memories; P*V B-frags from pre-transposed
// memT (both L2-resident). Fragment loads software-pipelined one step ahead.
__global__ void expert_kernel(const u16* __restrict__ mem_q,
                              const u16* __restrict__ mem_bf,
                              const u16* __restrict__ memT,
                              const int* __restrict__ cnt,
                              const int* __restrict__ offsets,
                              const int* __restrict__ a_tok,
                              const float* __restrict__ a_w,
                              const int* __restrict__ a_slot,
                              u16* __restrict__ mem_out) {
  int e = blockIdx.y;
  int count = cnt[e];
  int tile0 = blockIdx.x * 16;
  if (tile0 >= count) return;                 // uniform exit
  int nvalid = count - tile0; if (nvalid > 16) nvalid = 16;
  int base = offsets[e] + tile0;

  __shared__ __align__(16) u16 Qs[16 * 520];   // Q tile, padded rows
  __shared__ __align__(16) u16 Ps[16 * 136];   // softmax probs (bf16)
  __shared__ float Ss[16 * 132];               // raw scores
  __shared__ float rowM[16], rowL[16], rowA[16], rowW[16];
  __shared__ int   rowTok[16], rowSlot[16];

  int tid = threadIdx.x;
  int wave = tid >> 5, lane = tid & 31;
  int hi8 = (lane & 16) ? 8 : 0;

  if (tid < 16) {
    int r = tid, tok = 0, slot = -1; float w = 0.f;
    if (r < nvalid) { tok = a_tok[base + r]; w = a_w[base + r]; slot = a_slot[base + r]; }
    rowTok[r] = tok; rowW[r] = w; rowSlot[r] = slot;
    rowM[r] = -1e30f; rowL[r] = 0.f;
  }
  __syncthreads();

  // stage Q tile: 16 rows x 512 bf16 (async global->LDS)
  {
    int r = tid >> 3;
    const u16* src = mem_q + (size_t)rowTok[r] * DMEM;
#pragma unroll
    for (int i = 0; i < 8; i++) {
      int seg = (tid & 7) + i * 8;
      cp16(&Qs[r * 520 + seg * 8], src + seg * 8);
    }
  }
  async_fence();
  __syncthreads();

  const u16* Me  = mem_bf + (size_t)e * CAPS * DMEM;  // [slot][dmem]
  const u16* MeT = memT   + (size_t)e * DMEM * CAPS;  // [dmem][slot]

  v8f O[8];
#pragma unroll
  for (int nt = 0; nt < 8; nt++) O[nt] = (v8f){0,0,0,0,0,0,0,0};

  for (int c = 0; c < CAPS / CH; c++) {
    int s0 = c * CH + wave * 32 + (lane & 15);   // this wave's two slot tiles
    // QK^T: 16 tokens x 32 slots per wave, pipelined
    v8f sacc0 = (v8f){0,0,0,0,0,0,0,0};
    v8f sacc1 = (v8f){0,0,0,0,0,0,0,0};
    v16bf a  = load_frag(Qs, 520, lane & 15, 0, lane);
    v16bf b0 = load_frag(Me, DMEM, s0, 0, lane);
    v16bf b1 = load_frag(Me, DMEM, s0 + 16, 0, lane);
#pragma unroll
    for (int kk = 0; kk < 16; kk++) {
      v16bf an = a, b0n = b0, b1n = b1;
      if (kk < 15) {
        an  = load_frag(Qs, 520, lane & 15, (kk + 1) * 32, lane);
        b0n = load_frag(Me, DMEM, s0, (kk + 1) * 32, lane);
        b1n = load_frag(Me, DMEM, s0 + 16, (kk + 1) * 32, lane);
      }
      sacc0 = wmma_bf16(a, b0, sacc0);
      sacc1 = wmma_bf16(a, b1, sacc1);
      a = an; b0 = b0n; b1 = b1n;
    }
    __syncthreads();   // previous chunk's Ss/Ps consumers are done
#pragma unroll
    for (int v = 0; v < 8; v++) {
      int rr = (v + hi8) * 132 + wave * 32 + (lane & 15);
      Ss[rr]      = sacc0[v] * 0.044194173824159216f;  // 1/sqrt(512)
      Ss[rr + 16] = sacc1[v] * 0.044194173824159216f;
    }
    __syncthreads();

    // online softmax bookkeeping (wave 0, one lane per token row)
    if (wave == 0 && lane < 16) {
      int r = lane;
      float m0 = rowM[r], mx = -1e30f;
      for (int j = 0; j < CH; j++) mx = fmaxf(mx, Ss[r * 132 + j]);
      float mn = fmaxf(m0, mx);
      float alpha = __expf(m0 - mn);
      float s = 0.f;
      for (int j = 0; j < CH; j++) {
        float p = __expf(Ss[r * 132 + j] - mn);
        s += p;
        Ps[r * 136 + j] = f2bf(p);
      }
      rowL[r] = rowL[r] * alpha + s;
      rowM[r] = mn;
      rowA[r] = alpha;
    }
    __syncthreads();

    // rescale accumulators, then O += P * V over this wave's 128 dims
#pragma unroll
    for (int v = 0; v < 8; v++) {
      float al = rowA[v + hi8];
#pragma unroll
      for (int nt = 0; nt < 8; nt++) O[nt][v] *= al;
    }
#pragma unroll
    for (int kk = 0; kk < 4; kk++) {
      v16bf pa = load_frag(Ps, 136, lane & 15, kk * 32, lane);
      v16bf b = load_frag(MeT, CAPS, wave * 128 + (lane & 15),
                          c * CH + kk * 32, lane);
#pragma unroll
      for (int nt = 0; nt < 8; nt++) {
        v16bf bn = b;
        if (nt < 7)
          bn = load_frag(MeT, CAPS, wave * 128 + (nt + 1) * 16 + (lane & 15),
                         c * CH + kk * 32, lane);
        O[nt] = wmma_bf16(pa, b, O[nt]);
        b = bn;
      }
    }
  }
  __syncthreads();

  // epilogue: scale by top_w / l, store per-assignment output rows
#pragma unroll
  for (int v = 0; v < 8; v++) {
    int r = v + hi8;
    int slot = rowSlot[r];
    if (slot < 0) continue;
    float scl = rowW[r] / rowL[r];
#pragma unroll
    for (int nt = 0; nt < 8; nt++) {
      int col = wave * 128 + nt * 16 + (lane & 15);
      mem_out[(size_t)slot * DMEM + col] = f2bf(O[nt][v] * scl);
    }
  }
}

// ---------- combine the two expert outputs per token ----------
__global__ void comb_kernel(const u16* __restrict__ mem_out,
                            u16* __restrict__ comb) {
  int n = TOK * DMEM;
  for (int i = blockIdx.x * blockDim.x + threadIdx.x; i < n;
       i += gridDim.x * blockDim.x) {
    int t = i >> 9, m = i & 511;
    float a = bf2f(mem_out[(size_t)(2 * t) * DMEM + m]);
    float b = bf2f(mem_out[(size_t)(2 * t + 1) * DMEM + m]);
    comb[i] = f2bf(a + b);
  }
}

// ---------- launch ----------
extern "C" void kernel_launch(void* const* d_in, const int* in_sizes, int n_in,
                              void* d_out, int out_size, void* d_ws, size_t ws_size,
                              hipStream_t stream) {
  const float* hidden   = (const float*)d_in[0];
  const float* query    = (const float*)d_in[1];
  const float* router_w = (const float*)d_in[2];
  const float* memories = (const float*)d_in[3];
  const float* in_w     = (const float*)d_in[4];
  const float* in_b     = (const float*)d_in[5];
  const float* out_w    = (const float*)d_in[6];
  const float* out_b    = (const float*)d_in[7];
  float* out = (float*)d_out;
  float* loss_out = out + (size_t)TOK * DMOD;

  char* w = (char*)d_ws;
  // query_bf16 region (32 MB) is reused for mem_out + comb after the mem_q GEMM.
  u16* q_bf      = (u16*)(w + 0);
  u16* mem_out_bf= (u16*)(w + 0);
  u16* comb_bf   = (u16*)(w + 16777216);
  u16* inw_bf    = (u16*)(w + 33554432);
  u16* outw_bf   = (u16*)(w + 35651584);
  u16* mem_bf    = (u16*)(w + 37748736);
  u16* mem_q_bf  = (u16*)(w + 46137344);
  u16* memT_bf   = (u16*)(w + 54525952);
  int*   top_idx = (int*)(w + 62914560);
  float* top_w   = (float*)(w + 62980096);
  float* weights8= (float*)(w + 63045632);
  int*   a_tok   = (int*)(w + 63307776);
  float* a_w     = (float*)(w + 63373312);
  int*   a_slot  = (int*)(w + 63438848);
  int*   ctr     = (int*)(w + 63504384);
  int* cnt = ctr, *fill = ctr + 8, *disp = ctr + 16, *offs = ctr + 24;

  cvt_bf16_kernel<<<2048, 256, 0, stream>>>(query, q_bf, TOK * DMOD);
  cvt_bf16_kernel<<<512, 256, 0, stream>>>(in_w, inw_bf, DMEM * DMOD);
  cvt_bf16_kernel<<<512, 256, 0, stream>>>(out_w, outw_bf, DMOD * DMEM);
  cvt_bf16_kernel<<<1024, 256, 0, stream>>>(memories, mem_bf, NEXP * CAPS * DMEM);
  transpose_mem_kernel<<<dim3(DMEM / 32, CAPS / 32, NEXP), 256, 0, stream>>>(
      memories, memT_bf);
  init_counters_kernel<<<1, 64, 0, stream>>>(ctr);
  router_kernel<<<TOK / 8, 256, 0, stream>>>(hidden, router_w, top_idx, top_w,
                                             weights8, disp, cnt);
  scan_loss_kernel<<<1, 256, 0, stream>>>(weights8, disp, cnt, offs, loss_out);
  scatter_kernel<<<TOK / 256, 256, 0, stream>>>(top_idx, top_w, offs, fill,
                                                a_tok, a_w, a_slot);
  gemm_bf16_kernel<true><<<dim3(DMEM / 256, TOK / 32), 128, 0, stream>>>(
      q_bf, inw_bf, in_b, mem_q_bf, TOK, DMEM, DMOD);
  expert_kernel<<<dim3((2 * TOK) / 16, NEXP), 128, 0, stream>>>(
      mem_q_bf, mem_bf, memT_bf, cnt, offs, a_tok, a_w, a_slot, mem_out_bf);
  comb_kernel<<<4096, 256, 0, stream>>>(mem_out_bf, comb_bf);
  gemm_bf16_kernel<false><<<dim3(DMOD / 256, TOK / 32), 128, 0, stream>>>(
      comb_bf, outw_bf, out_b, out, TOK, DMOD, DMEM);
}
